// CategoricalGCNEncoder_67044439491023
// MI455X (gfx1250) — matrix operand order repaired and checked by reading, not compiled, not run on hardware
//
#include <hip/hip_runtime.h>
#include <hip/hip_bf16.h>

typedef _Float16 half_t;
typedef __attribute__((ext_vector_type(16))) _Float16 v16h;
typedef __attribute__((ext_vector_type(8)))  _Float16 v8h;
typedef __attribute__((ext_vector_type(8)))  float    v8f;

#define NNODES 100000
#define NEDGES 1600000
#define IN_DIM 128
#define HID    64
#define OUTF   32
#define MTILES (NNODES / 16)   // 6250

// ---------------- utility kernels ----------------
__global__ void fill_f32(float* __restrict__ p, float v, int n) {
    int i = blockIdx.x * blockDim.x + threadIdx.x;
    if (i < n) p[i] = v;
}

__global__ void deg_count(const int* __restrict__ dst, float* __restrict__ deg, int ne) {
    int i = blockIdx.x * blockDim.x + threadIdx.x;
    if (i < ne) atomicAdd(&deg[dst[i]], 1.0f);
}

__global__ void rsqrt_inplace(float* __restrict__ p, int n) {
    int i = blockIdx.x * blockDim.x + threadIdx.x;
    if (i < n) p[i] = rsqrtf(p[i]);
}

// embedding lookup + concat -> f16 feature matrix [N, 128]
__global__ void embed_kernel(const int* __restrict__ x_cat,
                             const float* __restrict__ emb,
                             half_t* __restrict__ x) {
    int t = blockIdx.x * blockDim.x + threadIdx.x;     // N*128 threads
    if (t >= NNODES * IN_DIM) return;
    int n = t >> 7;
    int col = t & 127;
    int f = col >> 4;          // field
    int e = col & 15;          // embedding element
    int idx = x_cat[n * 8 + f];
    x[t] = (half_t)emb[(f * 100 + idx) * 16 + e];
}

// ---------------- pack W[K,Ncols] (f32 row-major) into WMMA B-fragment layout ----------------
// Packed layout: Bp[((nt*KB + kb)*32 + lane)*16 + i]  holds  W[kb*32 + 16*(i>>3) + 8*(lane>>4) + (i&7)][nt*16 + (lane&15)]
// so each lane reads its 16-half fragment as one contiguous 32-byte vector.
__global__ void pack_b(const float* __restrict__ W, half_t* __restrict__ Bp, int K, int Ncols) {
    int t = blockIdx.x * blockDim.x + threadIdx.x;
    if (t >= K * Ncols) return;
    int i    = t & 15;
    int lane = (t >> 4) & 31;
    int rest = t >> 9;
    int kblocks = K >> 5;
    int kb = rest % kblocks;
    int nt = rest / kblocks;
    int k = kb * 32 + ((i >> 3) << 4) + ((lane >> 4) << 3) + (i & 7);
    int n = nt * 16 + (lane & 15);
    Bp[t] = (half_t)W[k * Ncols + n];
}

// ---------------- WMMA GEMM: H[M,NCOLS] = X[M,K] (f16) * packed-W (f16) ----------------
// One wave computes a full 16-row x NCOLS output stripe; A fragment reused across column tiles.
template <int K, int NCOLS, int WAVES>
__global__ __launch_bounds__(32 * WAVES)
void gemm_wmma(const half_t* __restrict__ X, const half_t* __restrict__ Bp,
               float* __restrict__ H) {
    constexpr int KB = K / 32;      // k-blocks
    constexpr int NT = NCOLS / 16;  // column tiles
    const int lane = threadIdx.x & 31;
    const int wv   = threadIdx.x >> 5;
    const int mt   = blockIdx.x * WAVES + wv;
    if (mt >= MTILES) return;
    const int row = mt * 16 + (lane & 15);   // A: lane carries matrix row M
    const int hs  = lane >> 4;               // lane-half selects K sub-block

    v8f acc[NT];
#pragma unroll
    for (int nt = 0; nt < NT; ++nt) acc[nt] = (v8f){};

    // A row base: elems 0-7 -> K=kk+(hs?8:0)+i ; elems 8-15 -> K=kk+16+(hs?8:0)+i
    const half_t* xr = X + (size_t)row * K + hs * 8;

#pragma unroll
    for (int kb = 0; kb < KB; ++kb) {
        v8h alo = *(const v8h*)(xr + kb * 32);
        v8h ahi = *(const v8h*)(xr + kb * 32 + 16);
        v16h a;
#pragma unroll
        for (int i = 0; i < 8; ++i) { a[i] = alo[i]; a[i + 8] = ahi[i]; }
#pragma unroll
        for (int nt = 0; nt < NT; ++nt) {
            const v16h b = *(const v16h*)(Bp + (size_t)(nt * KB + kb) * 512 + lane * 16);
            acc[nt] = __builtin_amdgcn_wmma_f32_16x16x32_f16(
                /*neg_a=*/false, a, /*neg_b=*/false, b,
                /*c_mod=*/(short)0, acc[nt], /*reuse_a=*/false, /*reuse_b=*/false);
        }
    }
    // D layout: VGPR r -> row mt*16 + r + (lane<16?0:8), col = nt*16 + (lane&15)
#pragma unroll
    for (int nt = 0; nt < NT; ++nt) {
        float* hp = H + (size_t)(mt * 16 + hs * 8) * NCOLS + nt * 16 + (lane & 15);
#pragma unroll
        for (int r = 0; r < 8; ++r) hp[(size_t)r * NCOLS] = acc[nt][r];
    }
}

// ---------------- edge scatter-add: one wave per edge, coalesced rows ----------------
__global__ void scatter_kernel(const int* __restrict__ src, const int* __restrict__ dst,
                               const float* __restrict__ dinv,
                               const float* __restrict__ h, float* __restrict__ agg,
                               int ne, int F) {
    int gid  = blockIdx.x * blockDim.x + threadIdx.x;
    int e    = gid >> 5;
    int lane = gid & 31;
    if (e >= ne) return;
    int s = src[e], d = dst[e];
    float w = dinv[s] * dinv[d];
    const float* hs = h + (size_t)s * F;
    float* ad = agg + (size_t)d * F;
    for (int j = lane; j < F; j += 32)
        atomicAdd(&ad[j], hs[j] * w);   // non-returning global_atomic_add_f32 (L2-resident)
}

// ---------------- post layer 1: self-loop + bias + ReLU + LayerNorm -> f16 ----------------
__global__ void post1_kernel(const float* __restrict__ agg, const float* __restrict__ h,
                             const float* __restrict__ dinv, const float* __restrict__ b,
                             const float* __restrict__ gamma, const float* __restrict__ beta,
                             half_t* __restrict__ xo) {
    int lane = threadIdx.x & 31;
    int n = (blockIdx.x * blockDim.x + threadIdx.x) >> 5;
    if (n >= NNODES) return;
    float di = dinv[n];
    float sw = di * di;                     // self-loop norm
    const float* ap = agg + (size_t)n * HID;
    const float* hp = h   + (size_t)n * HID;
    float v0 = ap[lane]      + hp[lane]      * sw + b[lane];
    float v1 = ap[lane + 32] + hp[lane + 32] * sw + b[lane + 32];
    v0 = fmaxf(v0, 0.0f); v1 = fmaxf(v1, 0.0f);           // ReLU
    float s = v0 + v1;
#pragma unroll
    for (int o = 16; o > 0; o >>= 1) s += __shfl_xor(s, o, 32);
    float mu = s * (1.0f / 64.0f);
    float d0 = v0 - mu, d1 = v1 - mu;
    float q = d0 * d0 + d1 * d1;
#pragma unroll
    for (int o = 16; o > 0; o >>= 1) q += __shfl_xor(q, o, 32);
    float rs = rsqrtf(q * (1.0f / 64.0f) + 1e-5f);
    half_t* xp = xo + (size_t)n * HID;
    xp[lane]      = (half_t)(d0 * rs * gamma[lane]      + beta[lane]);
    xp[lane + 32] = (half_t)(d1 * rs * gamma[lane + 32] + beta[lane + 32]);
}

// ---------------- post layer 2: self-loop + bias + LayerNorm, in place on d_out ----------------
__global__ void post2_kernel(float* __restrict__ out, const float* __restrict__ h,
                             const float* __restrict__ dinv, const float* __restrict__ b,
                             const float* __restrict__ gamma, const float* __restrict__ beta) {
    int lane = threadIdx.x & 31;
    int n = (blockIdx.x * blockDim.x + threadIdx.x) >> 5;
    if (n >= NNODES) return;
    float di = dinv[n];
    float sw = di * di;
    float* op = out + (size_t)n * OUTF;
    float v = op[lane] + h[(size_t)n * OUTF + lane] * sw + b[lane];
    float s = v;
#pragma unroll
    for (int o = 16; o > 0; o >>= 1) s += __shfl_xor(s, o, 32);
    float mu = s * (1.0f / 32.0f);
    float d0 = v - mu;
    float q = d0 * d0;
#pragma unroll
    for (int o = 16; o > 0; o >>= 1) q += __shfl_xor(q, o, 32);
    float rs = rsqrtf(q * (1.0f / 32.0f) + 1e-5f);
    op[lane] = d0 * rs * gamma[lane] + beta[lane];
}

// ---------------- launch ----------------
extern "C" void kernel_launch(void* const* d_in, const int* in_sizes, int n_in,
                              void* d_out, int out_size, void* d_ws, size_t ws_size,
                              hipStream_t stream) {
    (void)in_sizes; (void)n_in; (void)out_size; (void)ws_size;
    const int*   x_cat = (const int*)d_in[0];
    const int*   eidx  = (const int*)d_in[1];
    const float* emb   = (const float*)d_in[2];
    const float* W1    = (const float*)d_in[3];
    const float* b1    = (const float*)d_in[4];
    const float* W2    = (const float*)d_in[5];
    const float* b2    = (const float*)d_in[6];
    const float* g1    = (const float*)d_in[7];
    const float* be1   = (const float*)d_in[8];
    const float* g2    = (const float*)d_in[9];
    const float* be2   = (const float*)d_in[10];
    float* out = (float*)d_out;

    char* ws = (char*)d_ws;
    size_t off = 0;
    auto take = [&](size_t bytes) -> char* {
        char* p = ws + off;
        off = (off + bytes + 255) & ~(size_t)255;
        return p;
    };
    half_t* xh   = (half_t*)take((size_t)NNODES * IN_DIM * sizeof(half_t));
    half_t* w1p  = (half_t*)take((size_t)IN_DIM * HID * sizeof(half_t));
    half_t* w2p  = (half_t*)take((size_t)HID * OUTF * sizeof(half_t));
    float*  dinv = (float*)take((size_t)NNODES * sizeof(float));
    float*  h1   = (float*)take((size_t)NNODES * HID * sizeof(float));
    float*  agg1 = (float*)take((size_t)NNODES * HID * sizeof(float));
    half_t* x2h  = (half_t*)take((size_t)NNODES * HID * sizeof(half_t));
    float*  h2   = (float*)take((size_t)NNODES * OUTF * sizeof(float));

    const int* src = eidx;
    const int* dst = eidx + NEDGES;

    // degree (self-loops -> init 1.0), then rsqrt
    fill_f32<<<(NNODES + 255) / 256, 256, 0, stream>>>(dinv, 1.0f, NNODES);
    fill_f32<<<(NNODES * HID + 255) / 256, 256, 0, stream>>>(agg1, 0.0f, NNODES * HID);
    fill_f32<<<(NNODES * OUTF + 255) / 256, 256, 0, stream>>>(out, 0.0f, NNODES * OUTF);
    deg_count<<<(NEDGES + 255) / 256, 256, 0, stream>>>(dst, dinv, NEDGES);
    rsqrt_inplace<<<(NNODES + 255) / 256, 256, 0, stream>>>(dinv, NNODES);

    // pack weights into WMMA B-fragment layout (f16), build f16 feature matrix
    pack_b<<<(IN_DIM * HID + 255) / 256, 256, 0, stream>>>(W1, w1p, IN_DIM, HID);
    pack_b<<<(HID * OUTF + 255) / 256, 256, 0, stream>>>(W2, w2p, HID, OUTF);
    embed_kernel<<<(NNODES * IN_DIM + 255) / 256, 256, 0, stream>>>(x_cat, emb, xh);

    // layer 1: WMMA GEMM -> scatter -> ReLU+LN
    gemm_wmma<IN_DIM, HID, 8><<<(MTILES + 7) / 8, 256, 0, stream>>>(xh, w1p, h1);
    scatter_kernel<<<NEDGES / 8, 256, 0, stream>>>(src, dst, dinv, h1, agg1, NEDGES, HID);
    post1_kernel<<<(NNODES + 7) / 8, 256, 0, stream>>>(agg1, h1, dinv, b1, g1, be1, x2h);

    // layer 2: WMMA GEMM -> scatter -> LN (in place on d_out)
    gemm_wmma<HID, OUTF, 8><<<(MTILES + 7) / 8, 256, 0, stream>>>(x2h, w2p, h2);
    scatter_kernel<<<NEDGES / 8, 256, 0, stream>>>(src, dst, dinv, h2, out, NEDGES, OUTF);
    post2_kernel<<<(NNODES + 7) / 8, 256, 0, stream>>>(out, h2, dinv, b2, g2, be2);
}
